// CrfRnn3d_90795608638216
// MI455X (gfx1250) — matrix operand order repaired and checked by reading, not compiled, not run on hardware
//
#include <hip/hip_runtime.h>

#define NV 4096      // voxels = 16*16*16
#define LL 21        // labels
#define LP 32        // padded labels (two 16-row WMMA tiles)
#define SPLITK 8
#define KC (NV / SPLITK)   // 512 K-range per split
#define APITCH 520         // LDS pitch in halves: 512 + 8 -> conflict-free ds_load_b128

typedef __attribute__((ext_vector_type(16))) _Float16 v16h;
typedef __attribute__((ext_vector_type(8)))  float    v8f;

union FragU { v16h h; uint4 u[2]; };

#if defined(__has_builtin)
#if __has_builtin(__builtin_amdgcn_global_load_async_to_lds_b128) && \
    __has_builtin(__builtin_amdgcn_s_wait_asynccnt)
#define USE_ASYNC_LDS 1
#endif
#endif
#ifndef USE_ASYNC_LDS
#define USE_ASYNC_LDS 0
#endif

#if USE_ASYNC_LDS
// builtin signature: (v4i addrspace(1)*, v4i addrspace(3)*, imm offset, imm cpol)
typedef int v4i_t __attribute__((vector_size(16)));
typedef __attribute__((address_space(1))) v4i_t g_v4i;
typedef __attribute__((address_space(3))) v4i_t l_v4i;
#endif

// ---------------------------------------------------------------- build K_sp / K_bi (f16)
__global__ void build_kernels(const float* __restrict__ img,
                              _Float16* __restrict__ Ksp,
                              _Float16* __restrict__ Kbi) {
  size_t idx = (size_t)blockIdx.x * blockDim.x + threadIdx.x;
  int i = (int)(idx >> 12);
  int j = (int)(idx & (NV - 1));
  float dz = (float)((i >> 8) - (j >> 8));
  float dy = (float)(((i >> 4) & 15) - ((j >> 4) & 15));
  float dx = (float)((i & 15) - (j & 15));
  float d2 = dz * dz + dy * dy + dx * dx;           // GAMMA = 1
  Ksp[idx] = (_Float16)__expf(-0.5f * d2);
  const float invA2 = 1.0f / (67.0f * 67.0f);       // ALPHA = 67
  const float invB  = 1.0f / 3.0f;                  // BETA = 3
  float db = d2 * invA2;
#pragma unroll
  for (int c = 0; c < 3; ++c) {
    float di = (img[c * NV + i] - img[c * NV + j]) * invB;
    db += di * di;
  }
  Kbi[idx] = (_Float16)__expf(-0.5f * db);
}

// ---------------------------------------------------------------- 1 / column-sum (symmetric -> row sum)
__global__ void row_inv_norm(const _Float16* __restrict__ Ksp,
                             const _Float16* __restrict__ Kbi,
                             float* __restrict__ inv_sp,
                             float* __restrict__ inv_bi) {
  const _Float16* K = blockIdx.y ? Kbi : Ksp;
  float* out = blockIdx.y ? inv_bi : inv_sp;
  int row = blockIdx.x;
  float s = 0.f;
  for (int k = threadIdx.x; k < NV; k += blockDim.x)
    s += (float)K[(size_t)row * NV + k];
  __shared__ float red[256];
  red[threadIdx.x] = s;
  __syncthreads();
  for (int off = 128; off > 0; off >>= 1) {
    if (threadIdx.x < off) red[threadIdx.x] += red[threadIdx.x + off];
    __syncthreads();
  }
  if (threadIdx.x == 0) out[row] = 1.0f / red[0];
}

// ---------------------------------------------------------------- M = compat @ W (21x21, both)
__global__ void fold_weights(const float* __restrict__ compat,
                             const float* __restrict__ Wsp,
                             const float* __restrict__ Wbi,
                             float* __restrict__ Msp, float* __restrict__ Mbi) {
  int t = threadIdx.x;
  if (t >= LL * LL) return;
  int r = t / LL, c = t % LL;
  float ms = 0.f, mb = 0.f;
  for (int k = 0; k < LL; ++k) {
    ms += compat[r * LL + k] * Wsp[k * LL + c];
    mb += compat[r * LL + k] * Wbi[k * LL + c];
  }
  Msp[t] = ms;
  Mbi[t] = mb;
}

// ---------------------------------------------------------------- softmax over labels -> f16 q (padded to 32 rows)
__global__ void softmax_q(const float* __restrict__ cur, _Float16* __restrict__ qA) {
  int n = blockIdx.x * blockDim.x + threadIdx.x;
  if (n >= NV) return;
  float v[LL];
  float m = -1e30f;
#pragma unroll
  for (int l = 0; l < LL; ++l) { v[l] = cur[l * NV + n]; m = fmaxf(m, v[l]); }
  float s = 0.f;
#pragma unroll
  for (int l = 0; l < LL; ++l) { v[l] = __expf(v[l] - m); s += v[l]; }
  float inv = 1.0f / s;
#pragma unroll
  for (int l = 0; l < LL; ++l) qA[l * NV + n] = (_Float16)(v[l] * inv);
#pragma unroll
  for (int l = LL; l < LP; ++l) qA[l * NV + n] = (_Float16)0.f;
}

// ---------------------------------------------------------------- C(32xNV) = qA(32xNV) @ K(NVxNV), split-K partials
// A slab staged in LDS once per block (async-to-LDS when available);
// B fragments loaded via symmetry: B[k][n] == K[n][k] -> contiguous 16B lane loads.
__global__ void __launch_bounds__(256)
wmma_qk(const _Float16* __restrict__ qA,
        const _Float16* __restrict__ Ksp, const _Float16* __restrict__ Kbi,
        float* __restrict__ Psp, float* __restrict__ Pbi) {
  const _Float16* Kmat = blockIdx.z ? Kbi : Ksp;
  float* partial       = blockIdx.z ? Pbi : Psp;

  const int lane  = threadIdx.x & 31;
  const int wave  = threadIdx.x >> 5;                 // 0..7
  const int n0    = blockIdx.x * 128 + wave * 16;     // 16-col tile per wave
  const int kBase = blockIdx.y * KC;
  const int hs    = lane >> 4;                        // half-wave select
  const int lr    = lane & 15;

  // ---- stage A slice (32 rows x KC cols) into LDS, padded pitch ----
  __shared__ __align__(16) _Float16 ldsA[LP * APITCH];
#pragma unroll
  for (int i = 0; i < (LP * KC / 8) / 256; ++i) {     // 8 chunks of b128 per thread
    int c    = threadIdx.x + i * 256;                 // chunk id
    int row  = c >> 6;                                // 64 chunks per row
    int col8 = (c & 63) * 8;                          // halves
    const _Float16* src = qA + (size_t)row * NV + kBase + col8;
    _Float16* dst = &ldsA[row * APITCH + col8];
#if USE_ASYNC_LDS
    __builtin_amdgcn_global_load_async_to_lds_b128(
        (g_v4i*)(size_t)src, (l_v4i*)(unsigned)(size_t)dst, 0, 0);
#else
    *(uint4*)dst = *(const uint4*)src;
#endif
  }
#if USE_ASYNC_LDS
  __builtin_amdgcn_s_wait_asynccnt(0);
#endif
  __syncthreads();

  v8f acc0 = {}, acc1 = {};
  const _Float16* aRow0 = &ldsA[lr * APITCH];          // A rows 0..15 (in LDS)
  const _Float16* aRow1 = &ldsA[(16 + lr) * APITCH];   // A rows 16..31
  const _Float16* bBase = Kmat + (size_t)(n0 + lr) * NV; // K row n (== column n)

  for (int kb = 0; kb < KC; kb += 32) {
    FragU a0, a1, b;
    const _Float16* ap0 = aRow0 + kb + hs * 8;     // K=0..7 / 8..15
    a0.u[0] = *(const uint4*)(ap0);
    a0.u[1] = *(const uint4*)(ap0 + 16);           // K=16..23 / 24..31
    const _Float16* ap1 = aRow1 + kb + hs * 8;
    a1.u[0] = *(const uint4*)(ap1);
    a1.u[1] = *(const uint4*)(ap1 + 16);
    const _Float16* bp = bBase + kBase + kb + hs * 8; // K=0..7 / 8..15 of col n
    b.u[0] = *(const uint4*)(bp);
    b.u[1] = *(const uint4*)(bp + 16);           // K=16..23 / 24..31
    acc0 = __builtin_amdgcn_wmma_f32_16x16x32_f16(false, a0.h, false, b.h,
                                                  (short)0, acc0, false, false);
    acc1 = __builtin_amdgcn_wmma_f32_16x16x32_f16(false, a1.h, false, b.h,
                                                  (short)0, acc1, false, false);
  }
  float* out = partial + (size_t)blockIdx.y * (LP * NV);
#pragma unroll
  for (int g = 0; g < 8; ++g) {
    int m = g + hs * 8;                                  // C layout: VGPR g, halves
    out[(size_t)m * NV + n0 + lr]        = acc0[g];
    out[(size_t)(16 + m) * NV + n0 + lr] = acc1[g];
  }
}

// ---------------------------------------------------------------- sum split-K partials (deterministic)
__global__ void reduce_partials(const float* __restrict__ Psp, const float* __restrict__ Pbi,
                                float* __restrict__ Ssp, float* __restrict__ Sbi) {
  int idx = blockIdx.x * blockDim.x + threadIdx.x;   // 0 .. 2*LP*NV-1
  int which = idx / (LP * NV);
  int r = idx - which * (LP * NV);
  const float* P = which ? Pbi : Psp;
  float* S = which ? Sbi : Ssp;
  float s = 0.f;
#pragma unroll
  for (int sp = 0; sp < SPLITK; ++sp) s += P[(size_t)sp * LP * NV + r];
  S[r] = s;
}

// ---------------------------------------------------------------- cur = unary + inv_sp*Msp@Ssp + inv_bi*Mbi@Sbi
__global__ void combine(const float* __restrict__ Ssp, const float* __restrict__ Sbi,
                        const float* __restrict__ inv_sp, const float* __restrict__ inv_bi,
                        const float* __restrict__ Msp, const float* __restrict__ Mbi,
                        const float* __restrict__ unary, float* __restrict__ out) {
  int n = blockIdx.x * blockDim.x + threadIdx.x;
  int l = blockIdx.y;
  if (n >= NV) return;
  float as = 0.f, ab = 0.f;
#pragma unroll
  for (int l2 = 0; l2 < LL; ++l2) {
    as += Msp[l * LL + l2] * Ssp[l2 * NV + n];
    ab += Mbi[l * LL + l2] * Sbi[l2 * NV + n];
  }
  out[l * NV + n] = unary[l * NV + n] + inv_sp[n] * as + inv_bi[n] * ab;
}

// ----------------------------------------------------------------
extern "C" void kernel_launch(void* const* d_in, const int* in_sizes, int n_in,
                              void* d_out, int out_size, void* d_ws, size_t ws_size,
                              hipStream_t stream) {
  (void)in_sizes; (void)n_in; (void)out_size; (void)ws_size;
  const float* image  = (const float*)d_in[0];  // (1,3,16,16,16)
  const float* logits = (const float*)d_in[1];  // (1,21,16,16,16)
  const float* Wsp    = (const float*)d_in[2];  // (21,21)
  const float* Wbi    = (const float*)d_in[3];  // (21,21)
  const float* compat = (const float*)d_in[4];  // (21,21)
  float* outp = (float*)d_out;

  char* ws = (char*)d_ws;
  size_t off = 0;
  const size_t ALGN = 255;
  #define WS_ALLOC(ptr, type, bytes) type* ptr = (type*)(ws + off); off = (off + (bytes) + ALGN) & ~ALGN;

  WS_ALLOC(Ksp,   _Float16, (size_t)NV * NV * 2);          // 32 MB
  WS_ALLOC(Kbi,   _Float16, (size_t)NV * NV * 2);          // 32 MB
  WS_ALLOC(qA,    _Float16, (size_t)LP * NV * 2);
  WS_ALLOC(Psp,   float,    (size_t)SPLITK * LP * NV * 4); // 4 MB
  WS_ALLOC(Pbi,   float,    (size_t)SPLITK * LP * NV * 4); // 4 MB
  WS_ALLOC(Ssp,   float,    (size_t)LP * NV * 4);
  WS_ALLOC(Sbi,   float,    (size_t)LP * NV * 4);
  WS_ALLOC(invSp, float,    (size_t)NV * 4);
  WS_ALLOC(invBi, float,    (size_t)NV * 4);
  WS_ALLOC(Msp,   float,    (size_t)LL * LL * 4);
  WS_ALLOC(Mbi,   float,    (size_t)LL * LL * 4);
  WS_ALLOC(cur,   float,    (size_t)LL * NV * 4);
  #undef WS_ALLOC

  // Fixed structures (once per call)
  build_kernels<<<dim3((NV * (size_t)NV) / 256), dim3(256), 0, stream>>>(image, Ksp, Kbi);
  row_inv_norm<<<dim3(NV, 2), dim3(256), 0, stream>>>(Ksp, Kbi, invSp, invBi);
  fold_weights<<<dim3(1), dim3(512), 0, stream>>>(compat, Wsp, Wbi, Msp, Mbi);

  // 5 mean-field iterations
  for (int it = 0; it < 5; ++it) {
    const float* curIn = (it == 0) ? logits : cur;
    float* curOut = (it == 4) ? outp : cur;
    softmax_q<<<dim3(NV / 256), dim3(256), 0, stream>>>(curIn, qA);
    wmma_qk<<<dim3(NV / 128, SPLITK, 2), dim3(256), 0, stream>>>(qA, Ksp, Kbi, Psp, Pbi);
    reduce_partials<<<dim3(2 * LP * NV / 256), dim3(256), 0, stream>>>(Psp, Pbi, Ssp, Sbi);
    combine<<<dim3(NV / 256, LL), dim3(256), 0, stream>>>(Ssp, Sbi, invSp, invBi,
                                                          Msp, Mbi, logits, curOut);
  }
}
